// EMDLossWithProjection_17678085390413
// MI455X (gfx1250) — compile-verified, matrix operand order checked by reading, not run on hardware
//
#include <hip/hip_runtime.h>
#include <hip/hip_bf16.h>

// ---------------------------------------------------------------------------
// Types for CDNA5 WMMA (wave32): D(f32 16x16) = A(bf16 16x32) x B(bf16 32x16) + C
// ---------------------------------------------------------------------------
typedef __attribute__((ext_vector_type(16))) __bf16 v16bf;
typedef __attribute__((ext_vector_type(8)))  __bf16 v8bf;
typedef __attribute__((ext_vector_type(4)))  __bf16 v4bf;
typedef __attribute__((ext_vector_type(8)))  float  v8f;
typedef __attribute__((ext_vector_type(4)))  float  v4f;

union Frag {          // one A or B operand: 16 bf16 per lane = two 16B chunks
    v16bf v;
    v8bf  h[2];
};

#define EMD_EPS 1e-6f

// ---------------------------------------------------------------------------
// CDNA5 async global->LDS copy (ASYNCcnt-tracked, no VGPR round trip).
// INST_OFFSET is added to BOTH the global and the LDS address (ISA 08 §4.4),
// so one base pair + immediate offsets moves a contiguous 64B slice.
// ---------------------------------------------------------------------------
__device__ __forceinline__ void async_copy_64B(unsigned ldsOff,
                                               unsigned long long gaddr) {
    asm volatile("global_load_async_to_lds_b128 %0, %1, off"
                 :: "v"(ldsOff), "v"(gaddr) : "memory");
    asm volatile("global_load_async_to_lds_b128 %0, %1, off offset:16"
                 :: "v"(ldsOff), "v"(gaddr) : "memory");
    asm volatile("global_load_async_to_lds_b128 %0, %1, off offset:32"
                 :: "v"(ldsOff), "v"(gaddr) : "memory");
    asm volatile("global_load_async_to_lds_b128 %0, %1, off offset:48"
                 :: "v"(ldsOff), "v"(gaddr) : "memory");
}

__device__ __forceinline__ void wait_async0() {
#if __has_builtin(__builtin_amdgcn_s_wait_asynccnt)
    __builtin_amdgcn_s_wait_asynccnt(0);
#else
    asm volatile("s_wait_asynccnt 0x0" ::: "memory");
#endif
}

// ---------------------------------------------------------------------------
// fp32 -> bf16 bulk convert (vectorized x4)
// ---------------------------------------------------------------------------
__global__ __launch_bounds__(256) void k_cvt(const float* __restrict__ X,
                                             __bf16* __restrict__ Y, long n) {
    long i = (long)blockIdx.x * blockDim.x + threadIdx.x;
    long stride = (long)gridDim.x * blockDim.x;
    const v4f* x4 = (const v4f*)X;
    v4bf* y4 = (v4bf*)Y;
    long n4 = n >> 2;
    for (long k = i; k < n4; k += stride) {
        v4f v = x4[k];
        v4bf o;
        o.x = (__bf16)v.x; o.y = (__bf16)v.y;
        o.z = (__bf16)v.z; o.w = (__bf16)v.w;
        y4[k] = o;
    }
}

// ---------------------------------------------------------------------------
// teacher: fp32 -> bf16 convert fused with per-row mean-of-squares (t2)
// ---------------------------------------------------------------------------
__global__ __launch_bounds__(256) void k_cvt_row_sq(const float* __restrict__ X,
                                                    __bf16* __restrict__ Y,
                                                    float* __restrict__ t2, int V) {
    __shared__ float sv[256];
    int row = blockIdx.x, tid = threadIdx.x;
    const float* xr = X + (size_t)row * V;
    __bf16* yr = Y + (size_t)row * V;
    float ss = 0.f;
    for (int c = tid; c < V; c += 256) {
        float v = xr[c];
        ss += v * v;
        yr[c] = (__bf16)v;
    }
    sv[tid] = ss; __syncthreads();
    for (int s = 128; s > 0; s >>= 1) {
        if (tid < s) sv[tid] += sv[tid + s];
        __syncthreads();
    }
    if (tid == 0) t2[row] = sv[0] / (float)V;
}

// ---------------------------------------------------------------------------
// Big projection GEMM:  C[M,N](bf16) = A[M,K](bf16) * Bw[N,K](bf16)^T
// 128x128 workgroup tile, 4 waves (2x2), each wave 64x64 = 4x4 WMMA frags.
// Double-buffered LDS staging via async global->LDS, K step = 32.
// ---------------------------------------------------------------------------
__global__ __launch_bounds__(128) void k_gemm_proj(const __bf16* __restrict__ A,
                                                   const __bf16* __restrict__ Bw,
                                                   __bf16* __restrict__ C,
                                                   int M, int N, int K) {
    __shared__ __align__(16) __bf16 lA[2][128][32];
    __shared__ __align__(16) __bf16 lB[2][128][32];

    const int tid  = threadIdx.x;       // 0..127
    const int wave = tid >> 5;          // 0..3
    const int lane = tid & 31;
    const int half = lane >> 4;         // 0/1  (K-chunk selector)
    const int r    = lane & 15;         // row (A) / col (B) within 16
    const int m0 = blockIdx.y * 128;
    const int n0 = blockIdx.x * 128;
    const int wm = (wave & 1) * 64;     // wave's 64x64 sub-tile
    const int wn = (wave >> 1) * 64;

    v8f acc[4][4] = {};

    const __bf16* rowA = A  + (size_t)(m0 + tid) * K;   // this thread's A row
    const __bf16* rowB = Bw + (size_t)(n0 + tid) * K;   // this thread's B row

    auto loadTileAsync = [&](int buf, int k0) {
        unsigned aOff = (unsigned)(size_t)&lA[buf][tid][0];
        unsigned bOff = (unsigned)(size_t)&lB[buf][tid][0];
        const __bf16* ga = rowA + k0;
        const __bf16* gb = rowB + k0;
        if (k0 + 64 < K) {              // warm L2 for the k+2 stage
            __builtin_prefetch((const char*)ga + 128, 0, 2);
            __builtin_prefetch((const char*)gb + 128, 0, 2);
        }
        async_copy_64B(aOff, (unsigned long long)(size_t)ga);
        async_copy_64B(bOff, (unsigned long long)(size_t)gb);
    };

    auto compute = [&](int buf) {
        Frag a[4], b[4];
#pragma unroll
        for (int t = 0; t < 4; ++t) {
            const v8bf* pa = (const v8bf*)&lA[buf][wm + t * 16 + r][0];
            a[t].h[0] = pa[half];       // K = half*8 .. +7
            a[t].h[1] = pa[2 + half];   // K = 16+half*8 .. +7
            const v8bf* pb = (const v8bf*)&lB[buf][wn + t * 16 + r][0];
            b[t].h[0] = pb[half];
            b[t].h[1] = pb[2 + half];
        }
#pragma unroll
        for (int mi = 0; mi < 4; ++mi)
#pragma unroll
            for (int nj = 0; nj < 4; ++nj)
                acc[mi][nj] = __builtin_amdgcn_wmma_f32_16x16x32_bf16(
                    false, a[mi].v, false, b[nj].v,
                    (short)0, acc[mi][nj], false, false);
    };

    const int nk = K / 32;
    loadTileAsync(0, 0);
    wait_async0();                      // first tile resident
    __syncthreads();
    for (int kk = 0; kk < nk; ++kk) {
        int buf = kk & 1;
        if (kk + 1 < nk) loadTileAsync(buf ^ 1, (kk + 1) * 32);  // overlap
        compute(buf);
        wait_async0();                  // my async writes to buf^1 landed
        __syncthreads();                // everyone's writes visible
    }

    // Epilogue: C/D layout -> row = i + 8*half, col = r (per 16x16 sub-tile)
#pragma unroll
    for (int mi = 0; mi < 4; ++mi)
#pragma unroll
        for (int nj = 0; nj < 4; ++nj)
#pragma unroll
            for (int i = 0; i < 8; ++i) {
                int gm = m0 + wm + mi * 16 + i + 8 * half;
                int gn = n0 + wn + nj * 16 + r;
                C[(size_t)gm * N + gn] = (__bf16)acc[mi][nj][i];
            }
}

// ---------------------------------------------------------------------------
// per-row stats of aligned: s2 = mean(x^2), mask = (argmax(x) != 0)
// first-occurrence argmax semantics (tie -> lower index)
// ---------------------------------------------------------------------------
__global__ __launch_bounds__(256) void k_row_stats(const __bf16* __restrict__ X,
                                                   float* __restrict__ s2,
                                                   float* __restrict__ msk, int V) {
    __shared__ float sv[256];
    __shared__ float sm[256];
    __shared__ int   si[256];
    int row = blockIdx.x, tid = threadIdx.x;
    const __bf16* xr = X + (size_t)row * V;
    float ss = 0.f, best = -3.4e38f;
    int bidx = 0x7fffffff;
    for (int c = tid; c < V; c += 256) {
        float v = (float)xr[c];
        ss += v * v;
        if (v > best) { best = v; bidx = c; }
    }
    sv[tid] = ss; sm[tid] = best; si[tid] = bidx;
    __syncthreads();
    for (int s = 128; s > 0; s >>= 1) {
        if (tid < s) {
            sv[tid] += sv[tid + s];
            float v2 = sm[tid + s]; int i2 = si[tid + s];
            if (v2 > sm[tid] || (v2 == sm[tid] && i2 < si[tid])) {
                sm[tid] = v2; si[tid] = i2;
            }
        }
        __syncthreads();
    }
    if (tid == 0) {
        s2[row]  = sv[0] / (float)V;
        msk[row] = (si[0] != 0) ? 1.f : 0.f;
    }
}

// ---------------------------------------------------------------------------
// Per-block ground distance: D[i][j] = s2[i] + t2[j] - 2 * <sb_i, tb_j>/V
// 1 workgroup per (b,n) block; 4 waves split K, bf16 WMMA, LDS merge.
// ---------------------------------------------------------------------------
__global__ __launch_bounds__(128) void k_block_D(const __bf16* __restrict__ AL,
                                                 const __bf16* __restrict__ TB,
                                                 const float* __restrict__ s2,
                                                 const float* __restrict__ t2,
                                                 float* __restrict__ Dout, int V) {
    __shared__ float sD[64][64];
    const int tid = threadIdx.x, wave = tid >> 5, lane = tid & 31;
    const int half = lane >> 4, r = lane & 15;
    const size_t base = (size_t)blockIdx.x * 64;   // row = blk*64 + i
    const int kBeg = wave * (V / 4), kEnd = kBeg + V / 4;

    v8f acc[4][4] = {};
    for (int k = kBeg; k < kEnd; k += 32) {
        Frag a[4], b[4];
#pragma unroll
        for (int t = 0; t < 4; ++t) {
            const v8bf* pa = (const v8bf*)(AL + (base + t * 16 + r) * V + k);
            a[t].h[0] = pa[half];
            a[t].h[1] = pa[2 + half];
            const v8bf* pb = (const v8bf*)(TB + (base + t * 16 + r) * V + k);
            b[t].h[0] = pb[half];
            b[t].h[1] = pb[2 + half];
        }
#pragma unroll
        for (int mi = 0; mi < 4; ++mi)
#pragma unroll
            for (int nj = 0; nj < 4; ++nj)
                acc[mi][nj] = __builtin_amdgcn_wmma_f32_16x16x32_bf16(
                    false, a[mi].v, false, b[nj].v,
                    (short)0, acc[mi][nj], false, false);
    }

    for (int idx = tid; idx < 4096; idx += 128) ((float*)sD)[idx] = 0.f;
    __syncthreads();
    for (int w = 0; w < 4; ++w) {       // deterministic wave-sequential merge
        if (wave == w) {
#pragma unroll
            for (int mi = 0; mi < 4; ++mi)
#pragma unroll
                for (int nj = 0; nj < 4; ++nj)
#pragma unroll
                    for (int i = 0; i < 8; ++i)
                        sD[mi * 16 + i + 8 * half][nj * 16 + r] += acc[mi][nj][i];
        }
        __syncthreads();
    }

    const float invV = 1.f / (float)V;
    for (int idx = tid; idx < 4096; idx += 128) {
        int i = idx >> 6, j = idx & 63;
        Dout[(size_t)blockIdx.x * 4096 + idx] =
            s2[base + i] + t2[base + j] - 2.f * sD[i][j] * invV;
    }
}

// ---------------------------------------------------------------------------
// Per-block EMD flow math. 64 threads (thread i = row i).
// ---------------------------------------------------------------------------
__global__ __launch_bounds__(64) void k_emd_block(const float* __restrict__ Dall,
                                                  const float* __restrict__ msk,
                                                  float* __restrict__ partial,
                                                  float invTemp) {
    __shared__ float F[64][65];
    __shared__ float red[64];
    __shared__ float w[64];
    const int blk = blockIdx.x, i = threadIdx.x;
    const float* Db = Dall + (size_t)blk * 4096;

    auto rsum = [&](float v) -> float {
        red[i] = v; __syncthreads();
        for (int s = 32; s > 0; s >>= 1) {
            if (i < s) red[i] += red[i + s];
            __syncthreads();
        }
        float rr = red[0]; __syncthreads();
        return rr;
    };
    auto rmax = [&](float v) -> float {
        red[i] = v; __syncthreads();
        for (int s = 32; s > 0; s >>= 1) {
            if (i < s) red[i] = fmaxf(red[i], red[i + s]);
            __syncthreads();
        }
        float rr = red[0]; __syncthreads();
        return rr;
    };
    auto flow = [&]() {  // F from weights only (w_T == w_S == w)
        float wsum = rsum(w[i]);                 // total_flow = min(ws, ws)
        float F0 = wsum * (1.f / 4096.f);
        float rowsum = 64.f * F0;
        float fr = F0 * w[i] / fmaxf(rowsum, EMD_EPS);
        float colsum = rsum(fr);                 // identical for every column
        float inv = 1.f / fmaxf(colsum, EMD_EPS);
        for (int j = 0; j < 64; ++j) F[i][j] = fmaxf(fr * w[j] * inv, 0.f);
        __syncthreads();
    };

    float mb = msk[blk * 64 + i];
    float msum = rsum(mb);
    w[i] = (1.f / 64.f) * mb / fmaxf(msum, EMD_EPS);
    __syncthreads();

    flow();

    float ct = 0.f;
    for (int j = 0; j < 64; ++j) ct += Db[i * 64 + j] * F[i][j];
    ct /= fmaxf(w[i], EMD_EPS);
    float csum = rsum(ct);
    float z = (csum / fmaxf(ct, EMD_EPS)) * invTemp;
    float zmax = rmax(z);
    float e = expf(z - zmax);
    float esum = rsum(e);
    w[i] = e / esum;
    __syncthreads();

    flow();

    float workr = 0.f, flr = 0.f;
    for (int j = 0; j < 64; ++j) {
        float f = F[i][j];
        workr += f * Db[i * 64 + j];
        flr += f;
    }
    float WK = rsum(workr);
    float TF = rsum(flr);
    if (i == 0)
        partial[blk] = (WK / fmaxf(TF, EMD_EPS)) * (msum * (1.f / 64.f));
}

// ---------------------------------------------------------------------------
// Final deterministic reduction -> scalar loss
// ---------------------------------------------------------------------------
__global__ __launch_bounds__(128) void k_final(const float* __restrict__ partial,
                                               float* __restrict__ out,
                                               int n, float scale) {
    __shared__ float red[128];
    int tid = threadIdx.x;
    float s = 0.f;
    for (int c = tid; c < n; c += 128) s += partial[c];
    red[tid] = s; __syncthreads();
    for (int k = 64; k > 0; k >>= 1) {
        if (tid < k) red[tid] += red[tid + k];
        __syncthreads();
    }
    if (tid == 0) out[0] = red[0] * scale;
}

// ---------------------------------------------------------------------------
extern "C" void kernel_launch(void* const* d_in, const int* in_sizes, int n_in,
                              void* d_out, int out_size, void* d_ws, size_t ws_size,
                              hipStream_t stream) {
    (void)in_sizes; (void)n_in; (void)out_size; (void)ws_size;
    constexpr int  Bb = 4, S = 2048, V = 8192, NB = 32;   // nb = S/64
    constexpr long M  = (long)Bb * S;                      // 8192 rows
    constexpr int  NBLK = Bb * NB;                         // 128 EMD blocks

    const float* student = (const float*)d_in[0];
    const float* teacher = (const float*)d_in[1];
    const float* W       = (const float*)d_in[2];

    char* ws = (char*)d_ws;
    size_t off = 0;
    auto alloc = [&](size_t bytes) -> void* {
        void* p = (void*)(ws + off);
        off += (bytes + 255) & ~((size_t)255);
        return p;
    };
    __bf16* Wb  = (__bf16*)alloc((size_t)V * V * 2);   // 128 MB (fits L2)
    __bf16* Sb  = (__bf16*)alloc((size_t)M * V * 2);   // 128 MB
    __bf16* Tb  = (__bf16*)alloc((size_t)M * V * 2);   // 128 MB
    __bf16* ALb = (__bf16*)alloc((size_t)M * V * 2);   // 128 MB (aligned)
    float* s2   = (float*)alloc((size_t)M * 4);
    float* t2   = (float*)alloc((size_t)M * 4);
    float* mskv = (float*)alloc((size_t)M * 4);
    float* Dws  = (float*)alloc((size_t)NBLK * 4096 * 4);
    float* part = (float*)alloc((size_t)NBLK * 4);

    // 1) fp32 -> bf16 conversions
    k_cvt<<<4096, 256, 0, stream>>>(W, Wb, (long)V * V);
    k_cvt<<<4096, 256, 0, stream>>>(student, Sb, M * V);
    k_cvt_row_sq<<<(int)M, 256, 0, stream>>>(teacher, Tb, t2, V);

    // 2) big projection GEMM: aligned = student @ W^T  (bf16 WMMA, f32 accum)
    dim3 gg(V / 128, (int)(M / 128));
    k_gemm_proj<<<gg, 128, 0, stream>>>(Sb, Wb, ALb, (int)M, V, V);

    // 3) per-row mask / s2 of aligned
    k_row_stats<<<(int)M, 256, 0, stream>>>(ALb, s2, mskv, V);

    // 4) per-block ground distance D via bf16 WMMA
    k_block_D<<<NBLK, 128, 0, stream>>>(ALb, Tb, s2, t2, Dws, V);

    // 5) per-block EMD flow math (temperature = 1.0)
    k_emd_block<<<NBLK, 64, 0, stream>>>(Dws, mskv, part, 1.0f);

    // 6) final reduction: mean over batch
    k_final<<<1, 128, 0, stream>>>(part, (float*)d_out, NBLK, 1.0f / (float)Bb);
}
